// WITRAN_2DPSGMU_Encoder_21208548507910
// MI455X (gfx1250) — compile-verified, matrix operand order checked by reading, not run on hardware
//
#include <hip/hip_runtime.h>

// ---------------- Problem constants (baked from reference setup) ----------------
#define BB   64        // batch
#define RR   32        // rows
#define CC   32        // cols
#define II   8         // input size
#define HH   256       // hidden
#define LL   63        // R + C - 1
#define MM   2048      // R * BB  (GEMM M)
#define NN   1536      // 6 * H   (GEMM N)
#define K1   544       // layer0 K: 2H + 32 (x padded 8->32)
#define K2   1024      // layer1 K: 2H + 2H
#define NLAY 2

typedef __attribute__((ext_vector_type(16))) __bf16 v16bf;
typedef __attribute__((ext_vector_type(8)))  __bf16 v8bf;
typedef __attribute__((ext_vector_type(8)))  float  v8f;

// ---------------- bf16 fragment loader (matches CDNA5 16-bit A/B layout) --------
// lane l (wave32): row/col = l&15, khalf = (l>>4)*8
// elements 0..7  = K = k0 + khalf + 0..7
// elements 8..15 = K = k0 + 16 + khalf + 0..7
__device__ __forceinline__ v16bf load_frag_bf16(const __bf16* base, int stride,
                                                int row, int k0, int khalf) {
    const __bf16* p0 = base + (size_t)row * stride + k0 + khalf;
    v8bf lo = *reinterpret_cast<const v8bf*>(p0);
    v8bf hi = *reinterpret_cast<const v8bf*>(p0 + 16);
    v16bf r;
#pragma unroll
    for (int i = 0; i < 8; ++i) { r[i] = lo[i]; r[8 + i] = hi[i]; }
    return r;
}

// ---------------- Weight conversion ----------------
__global__ __launch_bounds__(256) void conv_w1(const float* __restrict__ Wf,
                                               __bf16* __restrict__ W1b) {
    int idx = blockIdx.x * 256 + threadIdx.x;
    if (idx >= NN * K1) return;
    int g = idx / K1, k = idx - g * K1;
    float v = (k < (II + 2 * HH)) ? Wf[g * (II + 2 * HH) + k] : 0.0f;
    W1b[idx] = (__bf16)v;
}

__global__ __launch_bounds__(256) void conv_w2(const float* __restrict__ Wo,
                                               __bf16* __restrict__ W2b) {
    int idx = blockIdx.x * 256 + threadIdx.x;
    if (idx >= NN * K2) return;
    W2b[idx] = (__bf16)Wo[idx];
}

// ---------------- Per-layer state init + A(t=0) staging ----------------
__global__ __launch_bounds__(256) void init_state(float* __restrict__ h_row,
                                                  float* __restrict__ hc0,
                                                  __bf16* __restrict__ A, int lda,
                                                  const float* __restrict__ x_in,
                                                  const __bf16* __restrict__ a1bf,
                                                  int layer) {
    int m = blockIdx.x;      // 0..2047
    int h = threadIdx.x;     // 0..255
    h_row[m * HH + h] = 0.0f;
    hc0[m * HH + h]   = 0.0f;
    size_t arow = (size_t)m * lda;
    A[arow + h]      = (__bf16)0.0f;
    A[arow + HH + h] = (__bf16)0.0f;
    if (layer == 0) {
        if (h < 32) {
            int r = m >> 6, b = m & 63;
            // t=0: valid only when r==0, c=0
            float v = (h < II && r == 0) ? x_in[(size_t)b * RR * CC * II + h] : 0.0f;
            A[arow + 2 * HH + h] = (__bf16)v;
        }
    } else {
        A[arow + 2 * HH + h]      = a1bf[(size_t)m * LL * (2 * HH) + h];
        A[arow + 2 * HH + HH + h] = a1bf[(size_t)m * LL * (2 * HH) + HH + h];
    }
}

// ---------------- Gate GEMM: G[m,n] = sum_k A[m,k] * W[n,k] -------------------
// Block 256 threads = 8 wave32, block tile 128M x 128N.
// K staged through LDS in 32-wide chunks, double buffered:
//   A tile 128x32, B tile 128x32, row stride padded to 40 elems (80B) so the
//   16-lane b128 fragment reads are bank-conflict-free (20*r mod 64 permutes).
__global__ __launch_bounds__(256) void gate_gemm(const __bf16* __restrict__ A,
                                                 const __bf16* __restrict__ W,
                                                 float* __restrict__ G,
                                                 int K) {
    constexpr int SST = 40;                       // LDS row stride (elements)
    __shared__ __bf16 sm[2][2][128 * SST];        // [buf][0:A,1:B]  (40 KB)

    const int tid   = threadIdx.x;
    const int lane  = tid & 31;
    const int wave  = tid >> 5;
    const int wm    = wave >> 1;                  // 0..3 -> 32 M rows each
    const int wn    = wave & 1;                   // 0..1 -> 64 N cols each
    const int rc    = lane & 15;
    const int khalf = (lane >> 4) * 8;

    const int mblk = blockIdx.x * 128;
    const int nblk = blockIdx.y * 128;

    // staging: 128 rows x 32 cols per tile = 512 x 16B chunks; thread does
    // chunks tid and tid+256 of each tile -> rows row0 and row0+64
    const int row0 = tid >> 2;                    // 0..63
    const int colg = (tid & 3) * 8;               // element offset 0/8/16/24

    v8f zero = {0.f, 0.f, 0.f, 0.f, 0.f, 0.f, 0.f, 0.f};
    v8f acc[2][4];
#pragma unroll
    for (int i = 0; i < 2; ++i)
#pragma unroll
        for (int j = 0; j < 4; ++j) acc[i][j] = zero;

    const int nk = K / 32;

    auto gA = [&](int kk, int row) {
        return *reinterpret_cast<const v8bf*>(A + (size_t)(mblk + row) * K + kk + colg);
    };
    auto gB = [&](int kk, int row) {
        return *reinterpret_cast<const v8bf*>(W + (size_t)(nblk + row) * K + kk + colg);
    };

    // stage chunk 0 into buffer 0
    {
        v8bf pa0 = gA(0, row0), pa1 = gA(0, row0 + 64);
        v8bf pb0 = gB(0, row0), pb1 = gB(0, row0 + 64);
        *reinterpret_cast<v8bf*>(&sm[0][0][row0 * SST + colg])        = pa0;
        *reinterpret_cast<v8bf*>(&sm[0][0][(row0 + 64) * SST + colg]) = pa1;
        *reinterpret_cast<v8bf*>(&sm[0][1][row0 * SST + colg])        = pb0;
        *reinterpret_cast<v8bf*>(&sm[0][1][(row0 + 64) * SST + colg]) = pb1;
    }
    __syncthreads();

    for (int kt = 0; kt < nk; ++kt) {
        const int buf  = kt & 1;
        const bool more = (kt + 1) < nk;

        // prefetch next chunk into registers
        v8bf pa0, pa1, pb0, pb1;
        if (more) {
            int kk = (kt + 1) * 32;
            pa0 = gA(kk, row0); pa1 = gA(kk, row0 + 64);
            pb0 = gB(kk, row0); pb1 = gB(kk, row0 + 64);
        }

        // fragments from LDS + 8 WMMAs
        const __bf16* sA = sm[buf][0];
        const __bf16* sB = sm[buf][1];
        v16bf a0 = load_frag_bf16(sA, SST, wm * 32 + rc,      0, khalf);
        v16bf a1 = load_frag_bf16(sA, SST, wm * 32 + 16 + rc, 0, khalf);
#pragma unroll
        for (int j = 0; j < 4; ++j) {
            v16bf b = load_frag_bf16(sB, SST, wn * 64 + j * 16 + rc, 0, khalf);
            acc[0][j] = __builtin_amdgcn_wmma_f32_16x16x32_bf16(
                false, a0, false, b, (short)0, acc[0][j], false, false);
            acc[1][j] = __builtin_amdgcn_wmma_f32_16x16x32_bf16(
                false, a1, false, b, (short)0, acc[1][j], false, false);
        }

        if (more) {
            __bf16* dA = sm[buf ^ 1][0];
            __bf16* dB = sm[buf ^ 1][1];
            *reinterpret_cast<v8bf*>(&dA[row0 * SST + colg])        = pa0;
            *reinterpret_cast<v8bf*>(&dA[(row0 + 64) * SST + colg]) = pa1;
            *reinterpret_cast<v8bf*>(&dB[row0 * SST + colg])        = pb0;
            *reinterpret_cast<v8bf*>(&dB[(row0 + 64) * SST + colg]) = pb1;
            __syncthreads();
        }
    }

    // C/D layout: n = lane&15 ; m = vgpr + 8*(lane>>4)
    const int mhalf = (lane >> 4) * 8;
#pragma unroll
    for (int i = 0; i < 2; ++i)
#pragma unroll
        for (int j = 0; j < 4; ++j) {
            int n  = nblk + wn * 64 + j * 16 + rc;
            int mb = mblk + wm * 32 + i * 16 + mhalf;
#pragma unroll
            for (int v = 0; v < 8; ++v)
                G[(size_t)(mb + v) * NN + n] = acc[i][j][v];
        }
}

// ---------------- Fused pointwise update + output + next-step staging ----------
__device__ __forceinline__ float sigmoidf(float x) {
    return 1.0f / (1.0f + __expf(-x));
}

__global__ __launch_bounds__(256) void pointwise_update(
    const float* __restrict__ G, const float* __restrict__ bias,
    float* __restrict__ h_row,
    const float* __restrict__ hc_in, float* __restrict__ hc_out,
    __bf16* __restrict__ Anext, int lda,
    const float* __restrict__ x_in,       // layer0 raw input
    const __bf16* __restrict__ a1bf_src,  // layer1 x-source (layer0 output, bf16)
    __bf16* __restrict__ a1bf_dst,        // layer0 output dest (bf16)
    float* __restrict__ out_final,        // layer1 output dest (fp32, d_out)
    float* __restrict__ hidr, float* __restrict__ hidc,
    int t, int layer) {
    int m = blockIdx.x;      // 0..2047 ; m = r*BB + b
    int h = threadIdx.x;     // 0..255
    int r = m >> 6, b = m & 63;

    size_t gb = (size_t)m * NN;
    float g0 = G[gb + h];
    float g1 = G[gb + HH + h];
    float g2 = G[gb + 2 * HH + h];
    float g3 = G[gb + 3 * HH + h];
    float g4 = G[gb + 4 * HH + h];
    float g5 = G[gb + 5 * HH + h];
    if (r <= t && t < RR) {   // reference mask: (arange(R) <= t) & (t < R)
        g0 += bias[h];
        g1 += bias[HH + h];
        g2 += bias[2 * HH + h];
        g3 += bias[3 * HH + h];
        g4 += bias[4 * HH + h];
        g5 += bias[5 * HH + h];
    }
    float u_r = sigmoidf(g0), o_r = sigmoidf(g1);
    float u_c = sigmoidf(g2), o_c = sigmoidf(g3);
    float i_r = tanhf(g4),    i_c = tanhf(g5);

    float hr_old = h_row[(size_t)m * HH + h];
    float hc_old = hc_in[(size_t)m * HH + h];
    float hr = tanhf((1.0f - u_r) * hr_old + u_r * i_r) * o_r;
    float hc = tanhf((1.0f - u_c) * hc_old + u_c * i_c) * o_c;

    h_row[(size_t)m * HH + h] = hr;
    int rolled = ((r + 1) & (RR - 1)) * BB + b;   // roll(h_col, +1) along rows
    hc_out[(size_t)rolled * HH + h] = hc;

    // layer output (out = concat([h_row, h_col]) pre-roll)
    if (layer == 0) {
        a1bf_dst[((size_t)m * LL + t) * (2 * HH) + h]      = (__bf16)hr;
        a1bf_dst[((size_t)m * LL + t) * (2 * HH) + HH + h] = (__bf16)hc;
    } else {
        out_final[((size_t)m * LL + t) * (2 * HH) + h]      = hr;
        out_final[((size_t)m * LL + t) * (2 * HH) + HH + h] = hc;
    }

    // hidden_row: saved at t>=C-1 from row r = t-(C-1)
    if (t >= CC - 1 && r == t - (CC - 1))
        hidr[(((size_t)b * NLAY + layer) * RR + (t - (CC - 1))) * HH + h] = hr;
    // hidden_col: saved at t>=R-1 from row R-1 (pre-roll)
    if (t >= RR - 1 && r == RR - 1)
        hidc[(((size_t)b * NLAY + layer) * CC + (t - (RR - 1))) * HH + h] = hc;

    // stage A for step t+1
    if (t + 1 < LL) {
        size_t am = (size_t)m * lda;
        Anext[am + h] = (__bf16)hr;
        Anext[(size_t)rolled * lda + HH + h] = (__bf16)hc;
        if (layer == 0) {
            if (h < 32) {
                int c = (t + 1) - r;
                float v = (h < II && c >= 0 && c < CC)
                              ? x_in[(((size_t)b * RR + r) * CC + c) * II + h]
                              : 0.0f;
                Anext[am + 2 * HH + h] = (__bf16)v;
            }
        } else {
            Anext[am + 2 * HH + h] =
                a1bf_src[((size_t)m * LL + (t + 1)) * (2 * HH) + h];
            Anext[am + 2 * HH + HH + h] =
                a1bf_src[((size_t)m * LL + (t + 1)) * (2 * HH) + HH + h];
        }
    }
}

// ---------------- Host launcher ----------------
extern "C" void kernel_launch(void* const* d_in, const int* in_sizes, int n_in,
                              void* d_out, int out_size, void* d_ws, size_t ws_size,
                              hipStream_t stream) {
    const float* inp  = (const float*)d_in[0];  // (B,R,C,I)
    const float* Wf   = (const float*)d_in[1];  // (1536, 520)
    const float* Wo   = (const float*)d_in[2];  // (1, 1536, 1024)
    const float* Bias = (const float*)d_in[3];  // (2, 1536)

    float* out = (float*)d_out;
    float* out_all = out;                                   // 2048*63*512
    float* hidr    = out + (size_t)MM * LL * (2 * HH);      // (B,2,R,H)
    float* hidc    = hidr + (size_t)BB * NLAY * RR * HH;    // (B,2,C,H)

    // workspace layout (256B-aligned slices)
    char* ws = (char*)d_ws;
    size_t off = 0;
    auto take = [&](size_t bytes) {
        char* p = ws + off;
        off += (bytes + 255) & ~(size_t)255;
        return p;
    };
    __bf16* A     = (__bf16*)take((size_t)MM * K2 * 2);          //  4.0 MB
    float*  gate  = (float*) take((size_t)MM * NN * 4);          // 12.6 MB
    float*  h_row = (float*) take((size_t)MM * HH * 4);          //  2.0 MB
    float*  hc0   = (float*) take((size_t)MM * HH * 4);          //  2.0 MB
    float*  hc1   = (float*) take((size_t)MM * HH * 4);          //  2.0 MB
    __bf16* W1b   = (__bf16*)take((size_t)NN * K1 * 2);          //  1.7 MB
    __bf16* W2b   = (__bf16*)take((size_t)NN * K2 * 2);          //  3.1 MB
    __bf16* a1bf  = (__bf16*)take((size_t)MM * LL * 2 * HH * 2); // 132 MB

    conv_w1<<<(NN * K1 + 255) / 256, 256, 0, stream>>>(Wf, W1b);
    conv_w2<<<(NN * K2 + 255) / 256, 256, 0, stream>>>(Wo, W2b);

    float* hcbuf[2] = {hc0, hc1};
    for (int layer = 0; layer < NLAY; ++layer) {
        const int     K   = (layer == 0) ? K1 : K2;
        const __bf16* Wb  = (layer == 0) ? W1b : W2b;
        init_state<<<MM, 256, 0, stream>>>(h_row, hc0, A, K, inp, a1bf, layer);
        for (int t = 0; t < LL; ++t) {
            gate_gemm<<<dim3(MM / 128, NN / 128), 256, 0, stream>>>(A, Wb, gate, K);
            pointwise_update<<<MM, 256, 0, stream>>>(
                gate, Bias + layer * NN, h_row,
                hcbuf[t & 1], hcbuf[(t + 1) & 1],
                A, K, inp, a1bf, a1bf, out_all, hidr, hidc, t, layer);
        }
    }
    (void)in_sizes; (void)n_in; (void)out_size; (void)ws_size;
}
